// self_attention_31499290149433
// MI455X (gfx1250) — matrix operand order repaired and checked by reading, not compiled, not run on hardware
//
#include <hip/hip_runtime.h>
#include <math.h>

// ---------------------------------------------------------------------------
// Problem constants (from reference): B=16, T=1200, C=1024, conv K=3
// ---------------------------------------------------------------------------
constexpr int Bn  = 16;
constexpr int Tn  = 1200;    // 75 tiles of 16
constexpr int Cn  = 1024;    // 64 tiles of 16, 32 K-steps of 32
constexpr int TPn = 1216;    // T padded to a multiple of 32 (zero-padded)
constexpr int TXn = Tn + 18; // x rows: row 0 and rows 1201..1217 are zero pads
constexpr int PADX = TXn - Tn;

typedef __bf16 bf16_t;
typedef __bf16 v8bf  __attribute__((ext_vector_type(8)));
typedef __bf16 v16bf __attribute__((ext_vector_type(16)));
typedef float  v8f   __attribute__((ext_vector_type(8)));

static __device__ inline v16bf cat8(v8bf lo, v8bf hi) {
  return __builtin_shufflevector(lo, hi, 0, 1, 2, 3, 4, 5, 6, 7,
                                 8, 9, 10, 11, 12, 13, 14, 15);
}

static __device__ inline v8f wmma_bf16(v16bf a, v16bf b, v8f c) {
  // D = A(16x32) * B(32x16) + C(16x16 f32)
  return __builtin_amdgcn_wmma_f32_16x16x32_bf16(
      /*neg_a=*/false, a, /*neg_b=*/false, b,
      /*c_mod=*/(short)0, c, /*reuse_a=*/false, /*reuse_b=*/false);
}

// ---------------------------------------------------------------------------
// Kernel 1: mask invalid timesteps (t >= q_len[b]) and cast f32 -> bf16.
// Writes into padded layout x[b][t+1][c].
// ---------------------------------------------------------------------------
__global__ __launch_bounds__(256) void k_prep_x(const float* __restrict__ q,
                                                const int* __restrict__ qlen,
                                                bf16_t* __restrict__ x) {
  long i = (long)blockIdx.x * 256 + threadIdx.x;
  const long total = (long)Bn * Tn * Cn;
  if (i >= total) return;
  int t = (int)((i / Cn) % Tn);
  int b = (int)(i / ((long)Tn * Cn));
  float v = (t < qlen[b]) ? q[i] : 0.0f;
  x[i + ((long)PADX * b + 1) * Cn] = (bf16_t)v;  // shift into padded rows
}

// ---------------------------------------------------------------------------
// Kernel 2: zero all pad regions: x pad rows, K pad rows, V pad columns.
// ---------------------------------------------------------------------------
__global__ __launch_bounds__(256) void k_zeropad(bf16_t* __restrict__ x,
                                                 bf16_t* __restrict__ Km,
                                                 bf16_t* __restrict__ Vm) {
  int i = blockIdx.x * 256 + threadIdx.x;
  const int nA = Bn * PADX * Cn;  // x: 18 pad rows per batch ({0} u [1201,1218))
  const int nB = Bn * 16 * Cn;    // K: 16 pad rows per batch
  const int nC = Bn * Cn * 16;    // V: 16 pad cols per (b, c) row
  if (i < nA) {
    int c = i % Cn, r = (i / Cn) % PADX, b = i / (PADX * Cn);
    int rowi = (r == 0) ? 0 : (Tn + r);  // 0, 1201..1217
    x[((long)b * TXn + rowi) * Cn + c] = (bf16_t)0.0f;
    return;
  }
  i -= nA;
  if (i < nB) {
    int c = i % Cn, r = (i / Cn) % 16, b = i / (16 * Cn);
    Km[((long)b * TPn + Tn + r) * Cn + c] = (bf16_t)0.0f;
    return;
  }
  i -= nB;
  if (i < nC) {
    int s = i % 16, c = (i / 16) % Cn, b = i / (16 * Cn);
    Vm[((long)b * Cn + c) * TPn + Tn + s] = (bf16_t)0.0f;
  }
}

// ---------------------------------------------------------------------------
// Kernel 3: repack conv weight [co][ci][k] f32 -> [k][co][ci] bf16.
// ---------------------------------------------------------------------------
__global__ __launch_bounds__(256) void k_prep_w(const float* __restrict__ w,
                                                bf16_t* __restrict__ wt) {
  int i = blockIdx.x * 256 + threadIdx.x;
  const int total = Cn * Cn * 3;
  if (i >= total) return;
  int k  = i % 3;
  int ci = (i / 3) % Cn;
  int co = i / (3 * Cn);
  wt[((long)k * Cn + co) * Cn + ci] = (bf16_t)w[i];
}

// ---------------------------------------------------------------------------
// Kernel 4: conv1d (pad=1) as 3 shifted GEMMs on WMMA. Branch-free inner
// loops (x is zero-padded). Register tiling: each wave owns 64(co) x 32(t):
// per K-step 2 B-loads + 4 A-loads feed 8 WMMAs (0.75 loads/WMMA).
// Epilogue guards t < Tn (keeps K/V zero pads intact).
// vmode==0: y -> [b][rowsb][co] bf16 (Q/K). vmode==1: y -> [b][co][TPn] (V).
// ---------------------------------------------------------------------------
__global__ __launch_bounds__(256) void k_conv(const bf16_t* __restrict__ x,
                                              const bf16_t* __restrict__ wt,
                                              const float* __restrict__ bias,
                                              bf16_t* __restrict__ out,
                                              int rowsb, int vmode) {
  const int lane = threadIdx.x & 31;
  const int wave = threadIdx.x >> 5;
  const int row  = lane & 15;   // A-row (co) / B-col (t) index for this lane
  const int hi   = lane >> 4;   // which K-half this lane holds
  const int t0   = blockIdx.x * 32;          // two 16-wide t tiles
  const int co0  = blockIdx.y * 512 + wave * 64;
  const int b    = blockIdx.z;

  const bf16_t* xb = x + (long)b * TXn * Cn;
  v8f acc[2][4] = {{{}, {}, {}, {}}, {{}, {}, {}, {}}};
  for (int k = 0; k < 3; ++k) {
    // padded row index: tsrc + 1 = t0 + row + (k-1) + 1
    const bf16_t* bptr = xb + (long)(t0 + row + k) * Cn + hi * 16;
    const bf16_t* aptr = wt + ((long)k * Cn + (co0 + row)) * Cn + hi * 16;
#pragma unroll 2
    for (int c = 0; c < Cn; c += 32) {
      v16bf b0 = *(const v16bf*)(bptr + c);
      v16bf b1 = *(const v16bf*)(bptr + (long)16 * Cn + c);
#pragma unroll
      for (int ti = 0; ti < 4; ++ti) {
        const bf16_t* ap = aptr + (long)ti * 16 * Cn + c;
        v16bf af = *(const v16bf*)(ap);
        acc[0][ti] = wmma_bf16(af, b0, acc[0][ti]);
        acc[1][ti] = wmma_bf16(af, b1, acc[1][ti]);
      }
    }
  }
  // C/D layout: lane gives N(=t)=row fixed, M(=co)=hi*8+r over the 8 VGPRs.
#pragma unroll
  for (int tj = 0; tj < 2; ++tj) {
    const int t = t0 + tj * 16 + row;
    if (t < Tn) {
#pragma unroll
      for (int ti = 0; ti < 4; ++ti) {
        const int cobase = co0 + ti * 16 + hi * 8;
        if (vmode == 0) {
          v8bf st;
          for (int r = 0; r < 8; ++r)
            st[r] = (bf16_t)(acc[tj][ti][r] + bias[cobase + r]);
          *(v8bf*)(out + ((long)b * rowsb + t) * Cn + cobase) = st;
        } else {
          for (int r = 0; r < 8; ++r)
            out[((long)b * Cn + cobase + r) * TPn + t] =
                (bf16_t)(acc[tj][ti][r] + bias[cobase + r]);
        }
      }
    }
  }
}

// ---------------------------------------------------------------------------
// Kernel 5: scores + softmax. One block per (b, 16-row t-tile). Each wave
// computes groups of 4 s-tiles (1 shared A-load + 4 B-loads + 4 WMMAs per
// K-step) over the zero-padded K, parks f32 scores in LDS [16][TPn], then a
// cooperative masked softmax writes bf16 P[b][t][s] (stride TPn, pads -> 0).
// ---------------------------------------------------------------------------
__global__ __launch_bounds__(256) void k_scores(const bf16_t* __restrict__ Q,
                                                const bf16_t* __restrict__ Kt,
                                                const int* __restrict__ qlen,
                                                bf16_t* __restrict__ P) {
  extern __shared__ float smem[];
  float* sc      = smem;              // 16 * TPn
  float* red     = smem + 16 * TPn;   // 16 * 16 partials
  float* rowstat = red + 256;         // 16

  const int b    = blockIdx.y;
  const int t0   = blockIdx.x * 16;
  const int lane = threadIdx.x & 31;
  const int wave = threadIdx.x >> 5;
  const int row  = lane & 15;
  const int hi   = lane >> 4;

  const bf16_t* aptr = Q + (((long)b * Tn + t0) + row) * Cn + hi * 16;
  const bf16_t* kb   = Kt + (long)b * TPn * Cn;

  const int nsg = TPn / 64;  // 19 groups of 4 s-tiles
  for (int sg = wave; sg < nsg; sg += 8) {
    const int s0 = sg * 64;
    const bf16_t* bbase = kb + (long)(s0 + row) * Cn + hi * 16;
    v8f acc[4] = {{}, {}, {}, {}};
#pragma unroll 2
    for (int c = 0; c < Cn; c += 32) {
      v16bf af = *(const v16bf*)(aptr + c);
#pragma unroll
      for (int ti = 0; ti < 4; ++ti) {
        v16bf bfv = *(const v16bf*)(bbase + (long)ti * 16 * Cn + c);
        acc[ti] = wmma_bf16(af, bfv, acc[ti]);
      }
    }
#pragma unroll
    for (int ti = 0; ti < 4; ++ti)
      for (int r = 0; r < 8; ++r)
        sc[(hi * 8 + r) * TPn + s0 + ti * 16 + row] = acc[ti][r];
  }
  __syncthreads();

  // Softmax: 16 threads per t-row, each strides over TPn/16 = 76 columns.
  const int   trow = threadIdx.x >> 4;
  const int   sub  = threadIdx.x & 15;
  const int   ql   = qlen[b];
  const float inv  = rsqrtf((float)ql);  // 1/sqrt(q_len)

  float m = -INFINITY;
  for (int j = sub; j < TPn; j += 16) {
    float v = (j < ql) ? sc[trow * TPn + j] * inv : -INFINITY;
    sc[trow * TPn + j] = v;
    m = fmaxf(m, v);
  }
  red[trow * 16 + sub] = m;
  __syncthreads();
  if (sub == 0) {
    float mm = red[trow * 16];
    for (int i = 1; i < 16; ++i) mm = fmaxf(mm, red[trow * 16 + i]);
    rowstat[trow] = mm;
  }
  __syncthreads();
  const float rm  = rowstat[trow];
  float       sum = 0.0f;
  for (int j = sub; j < TPn; j += 16) {
    float v = sc[trow * TPn + j];
    float e = (v == -INFINITY) ? 0.0f : __expf(v - rm);
    sc[trow * TPn + j] = e;
    sum += e;
  }
  __syncthreads();
  red[trow * 16 + sub] = sum;
  __syncthreads();
  if (sub == 0) {
    float ss = 0.0f;
    for (int i = 0; i < 16; ++i) ss += red[trow * 16 + i];
    rowstat[trow] = ss;
  }
  __syncthreads();
  const float rs     = rowstat[trow];
  const bool  tvalid = (t0 + trow) < ql;          // invalid t-rows unused later
  const float rinv   = (tvalid && rs > 0.0f) ? (1.0f / rs) : 0.0f;
  bf16_t* prow = P + ((long)b * Tn + t0 + trow) * TPn;
  for (int j = sub; j < TPn; j += 16)
    prow[j] = (bf16_t)(sc[trow * TPn + j] * rinv);
}

// ---------------------------------------------------------------------------
// Kernel 6: out = P * V^T, fused epilogue gamma*acc + query with validity
// select. Wave owns 16(t) x 64(c): per K-step 1 shared A-load (P) + 4
// B-loads (V) + 4 WMMAs; K loop over TPn/32 = 38 branch-free steps.
// ---------------------------------------------------------------------------
__global__ __launch_bounds__(256) void k_pv(const bf16_t* __restrict__ P,
                                            const bf16_t* __restrict__ V,
                                            const float* __restrict__ query,
                                            const int* __restrict__ qlen,
                                            const float* __restrict__ gamma,
                                            float* __restrict__ out) {
  const int b    = blockIdx.z;
  const int t0   = blockIdx.x * 16;
  const int wave = threadIdx.x >> 5;
  const int lane = threadIdx.x & 31;
  const int row  = lane & 15;
  const int hi   = lane >> 4;
  const int c0   = blockIdx.y * 512 + wave * 64;

  const bf16_t* prow  = P + ((long)b * Tn + t0 + row) * TPn + hi * 16;
  const bf16_t* vbase = V + ((long)b * Cn + c0 + row) * TPn + hi * 16;
  v8f acc[4] = {{}, {}, {}, {}};
#pragma unroll 2
  for (int s = 0; s < TPn; s += 32) {
    v16bf af = *(const v16bf*)(prow + s);
#pragma unroll
    for (int ti = 0; ti < 4; ++ti) {
      v16bf bfv = *(const v16bf*)(vbase + (long)ti * 16 * TPn + s);
      acc[ti] = wmma_bf16(af, bfv, acc[ti]);
    }
  }
  // C/D layout: N(=c)=row fixed per lane, M(=t)=hi*8+r over VGPRs.
  const float g  = gamma[0];
  const int   ql = qlen[b];
#pragma unroll
  for (int ti = 0; ti < 4; ++ti) {
    const int c = c0 + ti * 16 + row;
    for (int r = 0; r < 8; ++r) {
      const int  t   = t0 + hi * 8 + r;
      const long idx = ((long)b * Tn + t) * Cn + c;
      const float qv = query[idx];
      out[idx] = (t < ql) ? (g * acc[ti][r] + qv) : qv;
    }
  }
}

// ---------------------------------------------------------------------------
// Host launcher
// ---------------------------------------------------------------------------
extern "C" void kernel_launch(void* const* d_in, const int* in_sizes, int n_in,
                              void* d_out, int out_size, void* d_ws, size_t ws_size,
                              hipStream_t stream) {
  (void)in_sizes; (void)n_in; (void)out_size; (void)ws_size;
  const float* query = (const float*)d_in[0];
  const int*   qlen  = (const int*)d_in[1];
  const float* Wq    = (const float*)d_in[2];
  const float* bq    = (const float*)d_in[3];
  const float* Wk    = (const float*)d_in[4];
  const float* bk    = (const float*)d_in[5];
  const float* Wv    = (const float*)d_in[6];
  const float* bv    = (const float*)d_in[7];
  const float* gamma = (const float*)d_in[8];
  float*       out   = (float*)d_out;

  char*  ws  = (char*)d_ws;
  size_t off = 0;
  auto take = [&](size_t bytes) -> char* {
    char* p = ws + off;
    off = (off + bytes + 255) & ~(size_t)255;
    return p;
  };
  const size_t szX = (size_t)Bn * TXn * Cn * sizeof(bf16_t);  // padded x
  const size_t szW = (size_t)3 * Cn * Cn * sizeof(bf16_t);
  const size_t szQ = (size_t)Bn * Tn * Cn * sizeof(bf16_t);
  const size_t szK = (size_t)Bn * TPn * Cn * sizeof(bf16_t);  // padded rows
  const size_t szV = (size_t)Bn * Cn * TPn * sizeof(bf16_t);  // padded cols
  const size_t szP = (size_t)Bn * Tn * TPn * sizeof(bf16_t);
  bf16_t* xbt = (bf16_t*)take(szX);   // [b][TXn][c]
  bf16_t* wtq = (bf16_t*)take(szW);
  bf16_t* wtk = (bf16_t*)take(szW);
  bf16_t* wtv = (bf16_t*)take(szW);
  bf16_t* Qm  = (bf16_t*)take(szQ);   // [b][Tn][c]
  bf16_t* Km  = (bf16_t*)take(szK);   // [b][TPn][c]
  bf16_t* Vm  = (bf16_t*)take(szV);   // [b][c][TPn]
  bf16_t* Pm  = (bf16_t*)take(szP);   // [b][Tn][TPn]

  {
    long total = (long)Bn * Tn * Cn;
    int  grid  = (int)((total + 255) / 256);
    k_prep_x<<<grid, 256, 0, stream>>>(query, qlen, xbt);
  }
  {
    int total = Bn * PADX * Cn + Bn * 16 * Cn + Bn * Cn * 16;
    int grid  = (total + 255) / 256;
    k_zeropad<<<grid, 256, 0, stream>>>(xbt, Km, Vm);
  }
  {
    int total = Cn * Cn * 3;
    int grid  = (total + 255) / 256;
    k_prep_w<<<grid, 256, 0, stream>>>(Wq, wtq);
    k_prep_w<<<grid, 256, 0, stream>>>(Wk, wtk);
    k_prep_w<<<grid, 256, 0, stream>>>(Wv, wtv);
  }
  {
    dim3 g(TPn / 32, Cn / 512, Bn);   // 38 x 2 x 16 blocks, 8 waves each
    k_conv<<<g, 256, 0, stream>>>(xbt, wtq, bq, Qm, Tn, 0);
    k_conv<<<g, 256, 0, stream>>>(xbt, wtk, bk, Km, TPn, 0);
    k_conv<<<g, 256, 0, stream>>>(xbt, wtv, bv, Vm, 0, 1);
  }
  {
    dim3   g(Tn / 16, Bn);
    size_t smem = (size_t)(16 * TPn + 256 + 16) * sizeof(float);  // ~79 KB LDS
    k_scores<<<g, 256, smem, stream>>>(Qm, Km, qlen, Pm);
  }
  {
    dim3 g(Tn / 16, Cn / 512, Bn);
    k_pv<<<g, 256, 0, stream>>>(Pm, Vm, query, qlen, gamma, out);
  }
}
